// Attention_901943132827
// MI455X (gfx1250) — compile-verified, matrix operand order checked by reading, not run on hardware
//
#include <hip/hip_runtime.h>

// MI455X / gfx1250, wave32. bf16 WMMA (16x16x32) everywhere; f32 accumulate.
// Pipeline: cvt(x,Wqkv,Wproj) -> GEMM1(QKV, scatter per-head, V transposed)
//           -> fused per-(b,h) attention (scores in LDS, never in HBM)
//           -> GEMM2 (out proj, f32 + bias to d_out).
// GEMM: 128x256 block tile, 8 waves each owning 64x64 (16 WMMA / 32-K chunk),
// register-staged double-buffered LDS (48 KB), 1 barrier per K-chunk.

typedef __attribute__((ext_vector_type(16))) __bf16 v16bf;
typedef __attribute__((ext_vector_type(8)))  float  v8f;

static __device__ __forceinline__ unsigned short f2bf(float f) {
  unsigned int u = __float_as_uint(f);
  u += 0x7fffu + ((u >> 16) & 1u);          // round-to-nearest-even
  return (unsigned short)(u >> 16);
}

// K-offset of the bf16 pair held in fragment VGPR j for lane-half `half`
// (ISA 7.12.2, 16-bit A/B matrix 16x32 layout).
static __device__ __forceinline__ int kco(int j, int half) {
  return ((j >= 4) ? 16 : 0) + half * 8 + ((j & 3) << 1);
}

union Frag {
  v16bf v;
  unsigned int u[8];
};

// ---------------------------------------------------------------- conversion
__global__ __launch_bounds__(256) void cvt_f32_to_bf16_x4(
    const float* __restrict__ in, unsigned short* __restrict__ out, int n4) {
  int i = blockIdx.x * 256 + threadIdx.x;
  if (i >= n4) return;
  float4 f = ((const float4*)in)[i];
  uint2 o;
  o.x = (unsigned int)f2bf(f.x) | ((unsigned int)f2bf(f.y) << 16);
  o.y = (unsigned int)f2bf(f.z) | ((unsigned int)f2bf(f.w) << 16);
  ((uint2*)out)[i] = o;
}

// ---------------------------------------------------------------- GEMM (NT)
// C[M,N] = A[M,K] * Bw[N,K]^T + bias[N].  K % 32 == 0, M % 128 == 0, N % 256 == 0.
// MODE 0: write f32 to outF.  MODE 1: QKV epilogue -> scatter q/k (per head)
// and v transposed ([B,H,64,S]) as bf16.
template <int MODE>
__global__ __launch_bounds__(256) void gemm_bf16_nt(
    const unsigned short* __restrict__ A,
    const unsigned short* __restrict__ Bw,
    const float* __restrict__ bias,
    float* __restrict__ outF,
    unsigned short* __restrict__ qb,
    unsigned short* __restrict__ kb,
    unsigned short* __restrict__ vT,
    int M, int N, int K) {
  __shared__ __align__(16) unsigned short lA[2][128 * 32];  // 16 KB
  __shared__ __align__(16) unsigned short lB[2][256 * 32];  // 32 KB
  const int tid  = threadIdx.x;
  const int lane = tid & 31;
  const int wave = tid >> 5;
  const int lm = lane & 15, lh = lane >> 4;
  const int wm = wave & 1, wn = wave >> 1;      // 2 (M) x 4 (N) wave grid
  const int m0 = blockIdx.y * 128;
  const int n0 = blockIdx.x * 256;

  v8f acc[4][4];
  const v8f vzero = {0.f, 0.f, 0.f, 0.f, 0.f, 0.f, 0.f, 0.f};
#pragma unroll
  for (int mi = 0; mi < 4; ++mi)
#pragma unroll
    for (int ni = 0; ni < 4; ++ni) acc[mi][ni] = vzero;

  const int nk = K >> 5;
  uint4 ra[2], rb[4];                           // per-thread staged chunk

  auto load_chunk = [&](int kc) {
    const unsigned short* Ap = A  + (size_t)m0 * K + kc * 32;
    const unsigned short* Bp = Bw + (size_t)n0 * K + kc * 32;
#pragma unroll
    for (int i = 0; i < 2; ++i) {               // A: 128 rows x 4 uint4
      int c = tid + i * 256;
      int row = c >> 2, q4 = c & 3;
      ra[i] = *(const uint4*)(Ap + (size_t)row * K + q4 * 8);
    }
#pragma unroll
    for (int i = 0; i < 4; ++i) {               // B: 256 rows x 4 uint4
      int c = tid + i * 256;
      int row = c >> 2, q4 = c & 3;
      rb[i] = *(const uint4*)(Bp + (size_t)row * K + q4 * 8);
    }
  };
  auto store_chunk = [&](int buf) {
#pragma unroll
    for (int i = 0; i < 2; ++i) {
      int c = tid + i * 256;
      int row = c >> 2, q4 = c & 3;
      *(uint4*)&lA[buf][row * 32 + q4 * 8] = ra[i];
    }
#pragma unroll
    for (int i = 0; i < 4; ++i) {
      int c = tid + i * 256;
      int row = c >> 2, q4 = c & 3;
      *(uint4*)&lB[buf][row * 32 + q4 * 8] = rb[i];
    }
  };

  load_chunk(0);
  store_chunk(0);
  __syncthreads();

  for (int kc = 0; kc < nk; ++kc) {
    const int buf = kc & 1;
    if (kc + 1 < nk) load_chunk(kc + 1);        // overlap with compute below
    if (kc + 2 < nk) {                          // gfx1250 global_prefetch_b8
      int row = tid & 127;
      __builtin_prefetch(A + (size_t)(m0 + row) * K + (kc + 2) * 32, 0, 1);
      __builtin_prefetch(Bw + (size_t)(n0 + row + ((tid >> 7) << 7)) * K + (kc + 2) * 32, 0, 1);
    }

    Frag bfr[4];
#pragma unroll
    for (int ni = 0; ni < 4; ++ni)
#pragma unroll
      for (int j = 0; j < 8; ++j)
        bfr[ni].u[j] = *(const unsigned int*)&lB[buf][(wn * 64 + ni * 16 + lm) * 32 + kco(j, lh)];
#pragma unroll
    for (int mi = 0; mi < 4; ++mi) {
      Frag af;
#pragma unroll
      for (int j = 0; j < 8; ++j)
        af.u[j] = *(const unsigned int*)&lA[buf][(wm * 64 + mi * 16 + lm) * 32 + kco(j, lh)];
#pragma unroll
      for (int ni = 0; ni < 4; ++ni)
        acc[mi][ni] = __builtin_amdgcn_wmma_f32_16x16x32_bf16(
            false, af.v, false, bfr[ni].v, (short)0, acc[mi][ni], false, false);
    }

    if (kc + 1 < nk) store_chunk(buf ^ 1);
    __syncthreads();
  }

  // epilogue: C layout lane l, vgpr r -> (m = r + 8*(l/16), n = l%16)
#pragma unroll
  for (int mi = 0; mi < 4; ++mi) {
#pragma unroll
    for (int ni = 0; ni < 4; ++ni) {
#pragma unroll
      for (int r = 0; r < 8; ++r) {
        int m = m0 + wm * 64 + mi * 16 + r + 8 * lh;
        int n = n0 + wn * 64 + ni * 16 + lm;
        float val = acc[mi][ni][r] + bias[n];
        if (MODE == 0) {
          outF[(size_t)m * N + n] = val;
        } else {
          int which = n / 768;                  // 0=q 1=k 2=v
          int d = n - which * 768;
          int h = d >> 6, hd = d & 63;
          int b = m >> 9, s = m & 511;          // S = 512
          unsigned short bv = f2bf(val);
          size_t bhi = (size_t)(b * 12 + h);
          if (which == 0)      qb[(bhi * 512 + s) * 64 + hd] = bv;
          else if (which == 1) kb[(bhi * 512 + s) * 64 + hd] = bv;
          else                 vT[(bhi * 64 + hd) * 512 + s] = bv;  // transposed
        }
      }
    }
  }
}

// ---------------------------------------------------------------- attention
// One wave per (b,h, 16-row q tile). Scores [16x512] f32 in LDS, softmax,
// probs bf16 in LDS, P@V via WMMA. 1/rowsum folded into the output.
__global__ __launch_bounds__(32) void attn_fused(
    const unsigned short* __restrict__ qb,
    const unsigned short* __restrict__ kb,
    const unsigned short* __restrict__ vT,
    unsigned short* __restrict__ ao) {
  __shared__ __align__(16) float sc[16][512];            // 32 KB
  __shared__ __align__(16) unsigned short pb[16][512];   // 16 KB
  __shared__ float inv_s[16];
  const int qt = blockIdx.x;       // 0..31  (S/16 q tiles)
  const int bh = blockIdx.y;       // 0..383 (B*H)
  const int lane = threadIdx.x;
  const int lm = lane & 15, lh = lane >> 4;
  const unsigned short* qbase = qb + (size_t)bh * 512 * 64;
  const unsigned short* kbase = kb + (size_t)bh * 512 * 64;
  const unsigned short* vbase = vT + (size_t)bh * 64 * 512;
  const v8f vzero = {0.f, 0.f, 0.f, 0.f, 0.f, 0.f, 0.f, 0.f};

  Frag aQ[2];                      // Q tile 16x64 as two 16x32 A fragments
#pragma unroll
  for (int c = 0; c < 2; ++c)
#pragma unroll
    for (int j = 0; j < 8; ++j)
      aQ[c].u[j] = *(const unsigned int*)(qbase + (size_t)(qt * 16 + lm) * 64 + c * 32 + kco(j, lh));

  for (int kt = 0; kt < 32; ++kt) {            // 16 keys per tile
    Frag bK[2];
#pragma unroll
    for (int c = 0; c < 2; ++c)
#pragma unroll
      for (int j = 0; j < 8; ++j)
        bK[c].u[j] = *(const unsigned int*)(kbase + (size_t)(kt * 16 + lm) * 64 + c * 32 + kco(j, lh));
    v8f acc = vzero;
    acc = __builtin_amdgcn_wmma_f32_16x16x32_bf16(false, aQ[0].v, false, bK[0].v, (short)0, acc, false, false);
    acc = __builtin_amdgcn_wmma_f32_16x16x32_bf16(false, aQ[1].v, false, bK[1].v, (short)0, acc, false, false);
#pragma unroll
    for (int r = 0; r < 8; ++r)
      sc[r + 8 * lh][kt * 16 + lm] = acc[r] * 0.125f;    // SCALE = 64^-0.5
  }
  __syncthreads();

  {                                            // softmax: 2 lanes per row
    const int row = lane >> 1;
    const int c0 = (lane & 1) * 256;
    float mx = -3.4e38f;
    for (int c = 0; c < 256; ++c) mx = fmaxf(mx, sc[row][c0 + c]);
    mx = fmaxf(mx, __shfl_xor(mx, 1, 32));
    float sum = 0.f;
    for (int c = 0; c < 256; ++c) {
      float e = __expf(sc[row][c0 + c] - mx);
      sum += e;
      pb[row][c0 + c] = f2bf(e);               // unnormalized probs (bf16)
    }
    sum += __shfl_xor(sum, 1, 32);
    if ((lane & 1) == 0) inv_s[row] = 1.0f / sum;
  }
  __syncthreads();

  const int b = bh / 12, h = bh - (bh / 12) * 12;
#pragma unroll
  for (int nt = 0; nt < 4; ++nt) {             // 4 x 16 output dims
    v8f acc = vzero;
    for (int kt = 0; kt < 16; ++kt) {          // 32 keys per chunk
      Frag aP, bV;
#pragma unroll
      for (int j = 0; j < 8; ++j) {
        int kk = kt * 32 + kco(j, lh);
        aP.u[j] = *(const unsigned int*)&pb[lm][kk];
        bV.u[j] = *(const unsigned int*)(vbase + (size_t)(nt * 16 + lm) * 512 + kk);
      }
      acc = __builtin_amdgcn_wmma_f32_16x16x32_bf16(false, aP.v, false, bV.v, (short)0, acc, false, false);
    }
#pragma unroll
    for (int r = 0; r < 8; ++r) {
      int srow = qt * 16 + r + 8 * lh;
      float o = acc[r] * inv_s[r + 8 * lh];
      // [B,S,H,HD] == [M, 768] row-major for the output projection
      ao[((size_t)(b * 512 + srow) * 12 + h) * 64 + nt * 16 + lm] = f2bf(o);
    }
  }
}

// ---------------------------------------------------------------- launcher
extern "C" void kernel_launch(void* const* d_in, const int* in_sizes, int n_in,
                              void* d_out, int out_size, void* d_ws, size_t ws_size,
                              hipStream_t stream) {
  const float* x      = (const float*)d_in[0];   // [32,512,768]
  const float* qkv_w  = (const float*)d_in[1];   // [2304,768]
  const float* qkv_b  = (const float*)d_in[2];   // [2304]
  const float* proj_w = (const float*)d_in[3];   // [768,768]
  const float* proj_b = (const float*)d_in[4];   // [768]
  float* out = (float*)d_out;                    // [32,512,768] f32

  // Workspace layout (ushort elements), total ~130.5 MB.
  unsigned short* ws    = (unsigned short*)d_ws;
  unsigned short* xb    = ws;                    // 16384*768
  unsigned short* wqkv  = xb    + 12582912;      // 2304*768
  unsigned short* wproj = wqkv  + 1769472;       // 768*768
  unsigned short* qbuf  = wproj + 589824;        // [B,H,S,64]
  unsigned short* kbuf  = qbuf  + 12582912;      // [B,H,S,64]
  unsigned short* vTb   = kbuf  + 12582912;      // [B,H,64,S]
  unsigned short* aob   = vTb   + 12582912;      // [B,S,768]

  cvt_f32_to_bf16_x4<<<12288, 256, 0, stream>>>(x,      xb,    3145728);
  cvt_f32_to_bf16_x4<<<1728,  256, 0, stream>>>(qkv_w,  wqkv,  442368);
  cvt_f32_to_bf16_x4<<<576,   256, 0, stream>>>(proj_w, wproj, 147456);

  // QKV: M=16384, N=2304, K=768 -> scatter q/k/vT per head
  gemm_bf16_nt<1><<<dim3(9, 128), 256, 0, stream>>>(
      xb, wqkv, qkv_b, nullptr, qbuf, kbuf, vTb, 16384, 2304, 768);

  // Fused attention: grid (S/16, B*H), one wave per block
  attn_fused<<<dim3(32, 384), 32, 0, stream>>>(qbuf, kbuf, vTb, aob);

  // Output projection: M=16384, N=768, K=768 -> f32 + bias to d_out
  gemm_bf16_nt<0><<<dim3(3, 128), 256, 0, stream>>>(
      aob, wproj, proj_b, out, nullptr, nullptr, nullptr, 16384, 768, 768);

  (void)in_sizes; (void)n_in; (void)out_size; (void)ws_size;
}
